// AttentionBlock_39556648796228
// MI455X (gfx1250) — matrix operand order repaired and checked by reading, not compiled, not run on hardware
//
#include <hip/hip_runtime.h>

typedef __attribute__((ext_vector_type(16))) _Float16 v16h;
typedef __attribute__((ext_vector_type(8)))  _Float16 v8h;
typedef __attribute__((ext_vector_type(8)))  float    v8f;

__device__ __forceinline__ v8f wmma16(v16h a, v16h b, v8f c) {
  return __builtin_amdgcn_wmma_f32_16x16x32_f16(false, a, false, b, (short)0, c,
                                                false, false);
}

#define NROWS 8192   // B*H*W*D = 2*4096
#define CCH   128
#define NTOK  4096

// Fragment storage convention (wave32, CDNA5 ISA 7.12.2):
//  A-frag (16x32):  value X[m][k] lives at lane = m + 16*((k>>3)&1),
//                   elem = (k>>4)*8 + (k&7)          [k in 0..31]
//  B-frag (32x16):  value X[k][n] lives at lane = n + 16*((k>>3)&1),
//                   elem = (k>>4)*8 + (k&7)                    [k in 0..31]
// Each fragment = 32 lanes x 16 f16 = 1KB, loaded as one v16h (2x b128) per lane.

// ============ Kernel 0: weight swizzle (f32->f16 B-frags) + BN prep =========
__global__ __launch_bounds__(128) void prep_kernel(
    const float* __restrict__ Wq, const float* __restrict__ Wk,
    const float* __restrict__ Wv, const float* __restrict__ Wp,
    const float* __restrict__ gamma, const float* __restrict__ beta,
    const float* __restrict__ mmean, const float* __restrict__ mvar,
    _Float16* __restrict__ Wsw, float* __restrict__ bnscale,
    float* __restrict__ bnshift)
{
  int wave = threadIdx.x >> 5, lane = threadIdx.x & 31;
  {
    int c = threadIdx.x;  // 128 threads == 128 channels
    float s = gamma[c] * rsqrtf(mvar[c] + 1e-3f);
    bnscale[c] = s;
    bnshift[c] = beta[c] - mmean[c] * s;
  }
  const float* srcs[4] = {Wq, Wk, Wv, Wp};
  const float* W = srcs[wave];
  _Float16* dst = Wsw + (size_t)wave * CCH * CCH;
  int n = lane & 15, h = lane >> 4;
  for (int kk = 0; kk < 4; ++kk)
    for (int nt = 0; nt < 8; ++nt) {
      v16h f;
#pragma unroll
      for (int e = 0; e < 16; ++e)
        f[e] = (_Float16)W[(size_t)(kk * 32 + (e >> 3) * 16 + h * 8 + (e & 7)) * CCH + nt * 16 + n];
      *(v16h*)(dst + ((size_t)(kk * 8 + nt) * 32 + lane) * 16) = f;
    }
}

// ======================= Kernel 1: BN + QKV projections =====================
__global__ __launch_bounds__(128) void bn_qkv_kernel(
    const float* __restrict__ x, const float* __restrict__ bnscale,
    const float* __restrict__ bnshift, const _Float16* __restrict__ Wsw,
    const float* __restrict__ bq, const float* __restrict__ bk,
    const float* __restrict__ bv,
    float* __restrict__ xn_out, _Float16* __restrict__ Qsw,
    _Float16* __restrict__ Ksw, _Float16* __restrict__ Vsw)
{
  __shared__ __align__(32) _Float16 axn[4][4][32][16];  // per-wave xn A-frags
  int wave = threadIdx.x >> 5;
  int lane = threadIdx.x & 31;
  int tile = blockIdx.x * 4 + wave;        // 0..511
  int r0   = tile * 16;
  const float qscale = 0.08838834764831845f;  // 128^-0.5 folded into Q

  for (int idx = lane; idx < 16 * CCH; idx += 32) {
    int r = idx >> 7, c = idx & (CCH - 1);
    float val = fmaf(x[(size_t)(r0 + r) * CCH + c], bnscale[c], bnshift[c]);
    xn_out[(size_t)(r0 + r) * CCH + c] = val;
    int kk = c >> 5, lp = r + 16 * ((c >> 3) & 1), ep = ((c >> 4) & 1) * 8 + (c & 7);
    axn[wave][kk][lp][ep] = (_Float16)val;
  }
  __syncthreads();

  v16h afr[4];
#pragma unroll
  for (int kk = 0; kk < 4; ++kk) afr[kk] = *(const v16h*)&axn[wave][kk][lane][0];

  const float* bm[3] = {bq, bk, bv};
  int n = lane & 15, h = lane >> 4;
#pragma unroll
  for (int m3 = 0; m3 < 3; ++m3) {
    const _Float16* Wm = Wsw + (size_t)m3 * CCH * CCH;
    for (int nt = 0; nt < 8; ++nt) {
      v8f acc = {};
#pragma unroll
      for (int kk = 0; kk < 4; ++kk)
        acc = wmma16(afr[kk], *(const v16h*)(Wm + ((size_t)(kk * 8 + nt) * 32 + lane) * 16), acc);
      float bias = bm[m3][nt * 16 + n];
#pragma unroll
      for (int r = 0; r < 8; ++r) {
        int row = h * 8 + r;            // token within tile
        int c   = nt * 16 + n;          // output channel
        float v = acc[r] + bias;
        if (m3 == 0) {        // Q (pre-scaled) -> A-frag layout: m=row, k=c
          int kk = c >> 5, lp = row + 16 * ((c >> 3) & 1), ep = ((c >> 4) & 1) * 8 + (c & 7);
          Qsw[((size_t)(tile * 4 + kk) * 32 + lp) * 16 + ep] = (_Float16)(v * qscale);
        } else if (m3 == 1) { // K -> B(n,k)-frag: n=row(token), k=c
          int kk = c >> 5, lp = row + 16 * ((c >> 3) & 1), ep = ((c >> 4) & 1) * 8 + (c & 7);
          Ksw[((size_t)(tile * 4 + kk) * 32 + lp) * 16 + ep] = (_Float16)v;
        } else {              // V -> B(k,n)-frag: k=token(32-chunk), n=c
          int lp = n + 16 * ((row >> 3) & 1), ep = (tile & 1) * 8 + (row & 7);
          Vsw[((size_t)((tile >> 1) * 8 + nt) * 32 + lp) * 16 + ep] = (_Float16)v;
        }
      }
    }
  }
}

// ================ Kernel 2: flash attention + Wp + residual =================
// KV chunk = 64 keys: 32 WMMAs per softmax update / O-rescale.
struct __align__(32) WaveScratch {
  float    st[64][16];         // score chunk, TRANSPOSED [col][row]
  _Float16 pfrag[2][32][16];   // P as ready-to-load A-frags (k=0..31, 32..63)
  _Float16 ofrag[4][32][16];   // O as ready-to-load A-frags
};

__global__ __launch_bounds__(128) void flash_attn_kernel(
    const _Float16* __restrict__ Qsw, const _Float16* __restrict__ Ksw,
    const _Float16* __restrict__ Vsw, const float* __restrict__ xn,
    const _Float16* __restrict__ Wsw, const float* __restrict__ bp,
    float* __restrict__ out)
{
  __shared__ WaveScratch wsc[4];
  int wave = threadIdx.x >> 5, lane = threadIdx.x & 31;
  WaveScratch& S = wsc[wave];
  int tile  = blockIdx.x * 4 + wave;       // 0..511 (256 per batch)
  int batch = tile >> 8;
  int n = lane & 15, h = lane >> 4;

  v16h qa[4];
#pragma unroll
  for (int kk = 0; kk < 4; ++kk)
    qa[kk] = *(const v16h*)(Qsw + ((size_t)(tile * 4 + kk) * 32 + lane) * 16);

  v8f oacc[8];
#pragma unroll
  for (int i = 0; i < 8; ++i) oacc[i] = (v8f){};
  float m_reg = -1e30f, l_reg = 0.f;       // softmax state for row = lane&15

  const _Float16* Wpw = Wsw + (size_t)3 * CCH * CCH;

  for (int kv = 0; kv < NTOK; kv += 64) {
    int tt0 = batch * 256 + (kv >> 4);       // first of 4 16-token tiles
    // ---- S = Q @ K^T (16x64, scale pre-folded into Q) ----
    v8f sj[4];
#pragma unroll
    for (int j = 0; j < 4; ++j) sj[j] = (v8f){};
#pragma unroll
    for (int kk = 0; kk < 4; ++kk) {
#pragma unroll
      for (int j = 0; j < 4; ++j)
        sj[j] = wmma16(qa[kk],
                       *(const v16h*)(Ksw + ((size_t)((tt0 + j) * 4 + kk) * 32 + lane) * 16),
                       sj[j]);
    }
    // transposed store: 8 contiguous row-values per accumulator -> v8f stores
#pragma unroll
    for (int j = 0; j < 4; ++j)
      *(v8f*)&S.st[j * 16 + n][h * 8] = sj[j];
    __builtin_amdgcn_wave_barrier();
    // ---- online softmax: lane handles half of row (lane&15), 32 cols ----
    float alpha;
    {
      int row = n, c0 = h * 32;
      float mx = m_reg;
#pragma unroll
      for (int j = 0; j < 32; ++j) mx = fmaxf(mx, S.st[c0 + j][row]);
      mx = fmaxf(mx, __shfl_xor(mx, 16, 32));
      alpha = __expf(m_reg - mx);
      float lsum = 0.f;
#pragma unroll
      for (int g = 0; g < 4; ++g) {      // 8 probs packed per b128 LDS store
        int jj0 = c0 + g * 8;
        v8h pk;
#pragma unroll
        for (int e = 0; e < 8; ++e) {
          float pj = __expf(S.st[jj0 + e][row] - mx);
          lsum += pj;
          pk[e] = (_Float16)pj;
        }
        int fi = jj0 >> 5, kl = jj0 & 31;
        int lp = row + 16 * ((kl >> 3) & 1), ep0 = (kl >> 4) * 8;
        *(v8h*)&S.pfrag[fi][lp][ep0] = pk;  // A-frag layout, 16B aligned
      }
      lsum += __shfl_xor(lsum, 16, 32);
      m_reg = mx;
      l_reg = l_reg * alpha + lsum;
    }
    __builtin_amdgcn_wave_barrier();
    // ---- distribute alpha to C-layout rows via shuffle, rescale O ----
    float a8[8];
#pragma unroll
    for (int r = 0; r < 8; ++r) a8[r] = __shfl(alpha, h * 8 + r, 32);
#pragma unroll
    for (int nt = 0; nt < 8; ++nt)
#pragma unroll
      for (int r = 0; r < 8; ++r) oacc[nt][r] *= a8[r];
    // ---- O += P @ V_chunk (two 32-key halves) ----
    v16h pa0 = *(const v16h*)&S.pfrag[0][lane][0];
    v16h pa1 = *(const v16h*)&S.pfrag[1][lane][0];
    size_t vb = (size_t)(batch * 128 + (kv >> 5)) * 8;
#pragma unroll
    for (int nt = 0; nt < 8; ++nt)
      oacc[nt] = wmma16(pa0, *(const v16h*)(Vsw + ((vb + nt) * 32 + lane) * 16), oacc[nt]);
#pragma unroll
    for (int nt = 0; nt < 8; ++nt)
      oacc[nt] = wmma16(pa1, *(const v16h*)(Vsw + ((vb + 8 + nt) * 32 + lane) * 16), oacc[nt]);
  }

  // ---- normalize (1/l via shuffle) and stage O as A-frags in LDS ----
  float li = 1.f / l_reg;                   // valid for row = lane&15
  float linv[8];
#pragma unroll
  for (int r = 0; r < 8; ++r) linv[r] = __shfl(li, h * 8 + r, 32);
#pragma unroll
  for (int nt = 0; nt < 8; ++nt)
#pragma unroll
    for (int r = 0; r < 8; ++r) {
      int row = h * 8 + r, c = nt * 16 + n;
      int kk = c >> 5, lp = row + 16 * ((c >> 3) & 1), ep = ((c >> 4) & 1) * 8 + (c & 7);
      S.ofrag[kk][lp][ep] = (_Float16)(oacc[nt][r] * linv[r]);
    }
  __builtin_amdgcn_wave_barrier();

  // ---- out = xn + O @ Wp + bp ----
  v16h oa[4];
#pragma unroll
  for (int kk = 0; kk < 4; ++kk) oa[kk] = *(const v16h*)&S.ofrag[kk][lane][0];
  size_t base = (size_t)tile * 16 * CCH;
#pragma unroll
  for (int nt = 0; nt < 8; ++nt) {
    v8f acc = {};
#pragma unroll
    for (int kk = 0; kk < 4; ++kk)
      acc = wmma16(oa[kk], *(const v16h*)(Wpw + ((size_t)(kk * 8 + nt) * 32 + lane) * 16), acc);
    float bias = bp[nt * 16 + n];
#pragma unroll
    for (int r = 0; r < 8; ++r) {
      size_t idx = base + (size_t)(h * 8 + r) * CCH + nt * 16 + n;
      out[idx] = xn[idx] + acc[r] + bias;
    }
  }
}

extern "C" void kernel_launch(void* const* d_in, const int* in_sizes, int n_in,
                              void* d_out, int out_size, void* d_ws, size_t ws_size,
                              hipStream_t stream) {
  const float* x     = (const float*)d_in[0];
  const float* gamma = (const float*)d_in[1];
  const float* beta  = (const float*)d_in[2];
  const float* mmean = (const float*)d_in[3];
  const float* mvar  = (const float*)d_in[4];
  const float* Wq    = (const float*)d_in[5];
  const float* bq    = (const float*)d_in[6];
  const float* Wk    = (const float*)d_in[7];
  const float* bk    = (const float*)d_in[8];
  const float* Wv    = (const float*)d_in[9];
  const float* bv    = (const float*)d_in[10];
  const float* Wp    = (const float*)d_in[11];
  const float* bp    = (const float*)d_in[12];
  float* out = (float*)d_out;

  // Workspace: xn f32 4MB | Qsw 2MB | Ksw 2MB | Vsw 2MB | Wsw 128KB | BN 1KB
  char* ws = (char*)d_ws;
  float*    xn  = (float*)ws;
  _Float16* Qsw = (_Float16*)(ws + (size_t)NROWS * CCH * 4);
  _Float16* Ksw = Qsw + (size_t)NROWS * CCH;
  _Float16* Vsw = Ksw + (size_t)NROWS * CCH;
  _Float16* Wsw = Vsw + (size_t)NROWS * CCH;
  float*    bns = (float*)(Wsw + (size_t)4 * CCH * CCH);
  float*    bnh = bns + CCH;

  prep_kernel<<<1, 128, 0, stream>>>(Wq, Wk, Wv, Wp, gamma, beta, mmean, mvar,
                                     Wsw, bns, bnh);
  bn_qkv_kernel<<<128, 128, 0, stream>>>(x, bns, bnh, Wsw, bq, bk, bv,
                                         xn, Qsw, Ksw, Vsw);
  flash_attn_kernel<<<128, 128, 0, stream>>>(Qsw, Ksw, Vsw, xn, Wsw, bp, out);
}
